// GRUDecoder_25512105738642
// MI455X (gfx1250) — compile-verified, matrix operand order checked by reading, not run on hardware
//
#include <hip/hip_runtime.h>
#include <stdint.h>
#include <math.h>

// ---------- types ----------
typedef __attribute__((ext_vector_type(16))) __bf16   v16bf;
typedef __attribute__((ext_vector_type(8)))  float    v8f;
typedef __attribute__((ext_vector_type(4)))  uint32_t u32x4;

// ---------- constants (reference is fixed-shape) ----------
#define BATCH   256
#define IN_SZ   512          // x_t dimension (== OUT_DIM of the projection)
#define HID     1024
#define OUTD    512
#define SEQ     512
#define G3      (3 * HID)
#define KC      (IN_SZ + HID)   // 1536: concatenated [x|h] K-dim
#define NEG_SLOPE 0.04f

// ---------- helpers ----------
__device__ __forceinline__ uint16_t f2bf(float f) {
    uint32_t u = __float_as_uint(f);
    uint32_t r = u + 0x7FFFu + ((u >> 16) & 1u);   // round-to-nearest-even
    return (uint16_t)(r >> 16);
}

__device__ __forceinline__ float sigmoidf_fast(float x) {
    return 1.f / (1.f + __expf(-x));
}

__device__ __forceinline__ v16bf load_frag16(const uint16_t* p) {
    // ISA 16-bit fragment layout: 8 contiguous bf16 (this K-half) + 8 at +16
    union { u32x4 u[2]; v16bf v; } fr;
    fr.u[0] = *(const u32x4*)(p);
    fr.u[1] = *(const u32x4*)(p + 16);
    return fr.v;
}

#define WMMA_BF16(a, b, c) \
    __builtin_amdgcn_wmma_f32_16x16x32_bf16(false, (a), false, (b), (short)0, (c), false, false)

// ---------- generic bf16 WMMA GEMM:  out[m,n] = act( A[m,:]·W[n,:] + bias[n] )
// EPI: 1 = tanh (+bf16 copy), 2 = leaky_relu (+bf16 copy)
template <int EPI>
__global__ void __launch_bounds__(256, 4)
gemm_bf16_wmma(const uint16_t* __restrict__ A, int lda,
               const uint16_t* __restrict__ W, int ldw,
               const float* __restrict__ bias,
               int M, int N, int K,
               float* __restrict__ outF, int ldF,
               uint16_t* __restrict__ outB, int ldB)
{
    const int wave   = blockIdx.x * (blockDim.x >> 5) + (threadIdx.x >> 5);
    const int tilesN = N >> 4;
    const int total  = (M >> 4) * tilesN;
    if (wave >= total) return;                      // wave-uniform guard

    const int m0   = (wave / tilesN) << 4;
    const int n0   = (wave % tilesN) << 4;
    const int lane = threadIdx.x & 31;
    const int half = lane >> 4;
    const int lidx = lane & 15;

    const uint16_t* aRow = A + (size_t)(m0 + lidx) * lda + half * 8;
    const uint16_t* bRow = W + (size_t)(n0 + lidx) * ldw + half * 8;

    v8f acc = {0.f, 0.f, 0.f, 0.f, 0.f, 0.f, 0.f, 0.f};
    for (int kk = 0; kk < K; kk += 32) {
        v16bf a = load_frag16(aRow + kk);
        v16bf b = load_frag16(bRow + kk);
        acc = WMMA_BF16(a, b, acc);
    }

    const int   col   = n0 + lidx;
    const float bv    = bias[col];
    const int   rbase = m0 + half * 8;
#pragma unroll
    for (int v = 0; v < 8; ++v) {
        const int row = rbase + v;
        float val = acc[v] + bv;
        if (EPI == 1) val = tanhf(val);
        if (EPI == 2) val = (val >= 0.f) ? val : NEG_SLOPE * val;
        outF[(size_t)row * ldF + col] = val;
        outB[(size_t)row * ldB + col] = f2bf(val);
    }
}

// ---------- fully fused GRU step: gi + gh GEMMs + gates + state update ----------
// xhIn : (B, KC) bf16 = [x_t | h_t]      (read)
// wcat : (3H, KC) bf16, rows = [r|z|n] gate blocks, cols = [w_ih | w_hh]
// hOld : (B, H) f32 ; hNew : (B, H) f32
// xhOut: (B, KC) bf16 — h-section (cols IN_SZ..) written with new h
// NOTE __launch_bounds__(256, 2): relax occupancy target so the 4 accumulators
// plus the software-pipelined fragments fit in VGPRs (no scratch spills).
__global__ void __launch_bounds__(256, 2)
gru_step_wmma(const uint16_t* __restrict__ xhIn,
              const uint16_t* __restrict__ wcat,
              const float* __restrict__ b_ih,
              const float* __restrict__ b_hh,
              const float* __restrict__ hOld,
              float* __restrict__ hNew,
              uint16_t* __restrict__ xhOut)
{
    const int wave   = blockIdx.x * (blockDim.x >> 5) + (threadIdx.x >> 5);
    const int tilesN = HID >> 4;                       // 64
    if (wave >= (BATCH >> 4) * tilesN) return;

    const int m0   = (wave / tilesN) << 4;
    const int n0   = (wave % tilesN) << 4;
    const int lane = threadIdx.x & 31;
    const int half = lane >> 4;
    const int lidx = lane & 15;

    const uint16_t* aRow = xhIn + (size_t)(m0 + lidx) * KC + half * 8;
    const uint16_t* bR   = wcat + (size_t)(n0 + lidx) * KC + half * 8;
    const uint16_t* bZ   = bR + (size_t)HID * KC;
    const uint16_t* bN   = bR + (size_t)(2 * HID) * KC;

    v8f accR  = {0.f,0.f,0.f,0.f,0.f,0.f,0.f,0.f};
    v8f accZ  = accR;
    v8f accNi = accR;   // x-part of n gate  (inn)
    v8f accNh = accR;   // h-part of n gate  (hn)

    // K-chunks over x section: r, z, and inn all consume the same A fragment
    for (int kk = 0; kk < IN_SZ; kk += 32) {
        v16bf a = load_frag16(aRow + kk);
        accR  = WMMA_BF16(a, load_frag16(bR + kk), accR);
        accZ  = WMMA_BF16(a, load_frag16(bZ + kk), accZ);
        accNi = WMMA_BF16(a, load_frag16(bN + kk), accNi);
    }
    // K-chunks over h section: r, z, and hn share the A fragment
    for (int kk = IN_SZ; kk < KC; kk += 32) {
        v16bf a = load_frag16(aRow + kk);
        accR  = WMMA_BF16(a, load_frag16(bR + kk), accR);
        accZ  = WMMA_BF16(a, load_frag16(bZ + kk), accZ);
        accNh = WMMA_BF16(a, load_frag16(bN + kk), accNh);
    }

    const int   col = n0 + lidx;
    const float bRv = b_ih[col]           + b_hh[col];
    const float bZv = b_ih[HID + col]     + b_hh[HID + col];
    const float bNi = b_ih[2 * HID + col];
    const float bNh = b_hh[2 * HID + col];
    const int   rbase = m0 + half * 8;

#pragma unroll
    for (int v = 0; v < 8; ++v) {
        const int row = rbase + v;
        const float r = sigmoidf_fast(accR[v] + bRv);
        const float z = sigmoidf_fast(accZ[v] + bZv);
        const float n = tanhf(accNi[v] + bNi + r * (accNh[v] + bNh));
        const float h = (1.f - z) * n + z * hOld[(size_t)row * HID + col];
        hNew[(size_t)row * HID + col]          = h;
        xhOut[(size_t)row * KC + IN_SZ + col]  = f2bf(h);
    }
}

// ---------- weight packing: wcat[r, k] = bf16( k<IN_SZ ? w_ih[r,k] : w_hh[r,k-IN_SZ] )
__global__ void pack_wcat(const float* __restrict__ w_ih,   // (3H, IN_SZ)
                          const float* __restrict__ w_hh,   // (3H, HID)
                          uint16_t* __restrict__ wcat)      // (3H, KC)
{
    const int idx = blockIdx.x * blockDim.x + threadIdx.x;
    if (idx >= G3 * KC) return;
    const int row = idx / KC;
    const int k   = idx - row * KC;
    const float v = (k < IN_SZ) ? w_ih[(size_t)row * IN_SZ + k]
                                : w_hh[(size_t)row * HID + (k - IN_SZ)];
    wcat[idx] = f2bf(v);
}

// ---------- fp32 -> bf16 convert (contiguous) ----------
__global__ void cvt_f32_bf16(const float* __restrict__ in,
                             uint16_t* __restrict__ out, int n)
{
    const int i = blockIdx.x * blockDim.x + threadIdx.x;
    if (i < n) out[i] = f2bf(in[i]);
}

// ---------- host ----------
extern "C" void kernel_launch(void* const* d_in, const int* in_sizes, int n_in,
                              void* d_out, int out_size, void* d_ws, size_t ws_size,
                              hipStream_t stream)
{
    (void)in_sizes; (void)n_in; (void)out_size; (void)ws_size;

    const float* x        = (const float*)d_in[0];
    const float* fc_h0_w  = (const float*)d_in[1];
    const float* fc_h0_b  = (const float*)d_in[2];
    const float* w_ih     = (const float*)d_in[3];
    const float* w_hh     = (const float*)d_in[4];
    const float* b_ih     = (const float*)d_in[5];
    const float* b_hh     = (const float*)d_in[6];
    const float* fc_out_w = (const float*)d_in[7];
    const float* fc_out_b = (const float*)d_in[8];
    float* out = (float*)d_out;

    // workspace carve-up (256B-aligned slabs)
    size_t off = 0;
    auto take = [&](size_t bytes) -> void* {
        void* p = (char*)d_ws + off;
        off += (bytes + 255) & ~(size_t)255;
        return p;
    };
    uint16_t* wcat    = (uint16_t*)take((size_t)G3   * KC    * 2);  // packed [w_ih|w_hh]
    uint16_t* wh0_bf  = (uint16_t*)take((size_t)HID  * IN_SZ * 2);
    uint16_t* wout_bf = (uint16_t*)take((size_t)OUTD * HID   * 2);
    uint16_t* xin_bf  = (uint16_t*)take((size_t)BATCH * IN_SZ * 2);
    uint16_t* XH0     = (uint16_t*)take((size_t)BATCH * KC   * 2);  // [x|h] buf A
    uint16_t* XH1     = (uint16_t*)take((size_t)BATCH * KC   * 2);  // [x|h] buf B
    float*    hA      = (float*)   take((size_t)BATCH * HID  * 4);
    float*    hB      = (float*)   take((size_t)BATCH * HID  * 4);

    const int CT = 256;
    pack_wcat<<<(G3 * KC + CT - 1) / CT, CT, 0, stream>>>(w_ih, w_hh, wcat);
    cvt_f32_bf16<<<(HID * IN_SZ + CT - 1) / CT, CT, 0, stream>>>(fc_h0_w, wh0_bf, HID * IN_SZ);
    cvt_f32_bf16<<<(OUTD * HID + CT - 1) / CT, CT, 0, stream>>>(fc_out_w, wout_bf, OUTD * HID);
    cvt_f32_bf16<<<(BATCH * IN_SZ + CT - 1) / CT, CT, 0, stream>>>(x, xin_bf, BATCH * IN_SZ);

    auto blocksFor = [](int M, int N) {              // 8 waves/block, 1 tile/wave
        int tiles = (M >> 4) * (N >> 4);
        return (tiles + 7) / 8;
    };

    // h0 = tanh(x @ fc_h0_w.T + b) -> hA (fp32) + h-section of XH0 (bf16)
    gemm_bf16_wmma<1><<<blocksFor(BATCH, HID), 256, 0, stream>>>(
        xin_bf, IN_SZ, wh0_bf, IN_SZ, fc_h0_b,
        BATCH, HID, IN_SZ, hA, HID, XH0 + IN_SZ, KC);

    // x0 = leaky(h0 @ fc_out_w.T + b) -> out[:,0,:] + x-section of XH0
    gemm_bf16_wmma<2><<<blocksFor(BATCH, OUTD), 256, 0, stream>>>(
        XH0 + IN_SZ, KC, wout_bf, HID, fc_out_b,
        BATCH, OUTD, HID, out, SEQ * OUTD, XH0, KC);

    float*    hCur = hA;  float*    hNxt = hB;
    uint16_t* Xcur = XH0; uint16_t* Xnxt = XH1;

    const int stepBlocks = blocksFor(BATCH, HID);    // 128 blocks, 1024 waves
    const int projBlocks = blocksFor(BATCH, OUTD);   // 64 blocks, 512 waves

    for (int t = 1; t < SEQ; ++t) {
        // fused: gi + gh + gates -> hNxt (fp32), h-section of Xnxt (bf16)
        gru_step_wmma<<<stepBlocks, 256, 0, stream>>>(
            Xcur, wcat, b_ih, b_hh, hCur, hNxt, Xnxt);
        // x_t = leaky(h @ fc_out_w.T + b) -> out[:,t,:] + x-section of Xnxt
        gemm_bf16_wmma<2><<<projBlocks, 256, 0, stream>>>(
            Xnxt + IN_SZ, KC, wout_bf, HID, fc_out_b,
            BATCH, OUTD, HID, out + (size_t)t * OUTD, SEQ * OUTD, Xnxt, KC);
        float* th = hCur; hCur = hNxt; hNxt = th;
        uint16_t* tx = Xcur; Xcur = Xnxt; Xnxt = tx;
    }
}
